// NodeExtractionBasic_48284022341970
// MI455X (gfx1250) — compile-verified, matrix operand order checked by reading, not run on hardware
//
#include <hip/hip_runtime.h>
#include <hip/hip_bf16.h>
#include <stdint.h>

// ---------------- problem constants ----------------
constexpr int N  = 50000;
constexpr int E  = 800000;
constexpr int NO2 = 169;  // 13*13
constexpr int NO2P = 176; // padded to 11*16

typedef _Float16 v16h __attribute__((ext_vector_type(16)));
typedef _Float16 h8   __attribute__((ext_vector_type(8)));
typedef float    v8f  __attribute__((ext_vector_type(8)));

static __device__ inline v16h make16(h8 lo, h8 hi) {
  union { v16h v; h8 h[2]; } u; u.h[0] = lo; u.h[1] = hi; return u.v;
}

// A fragment for V_WMMA_F32_16X16X32_F16 from a row-major f16 buffer
// (LDS or global).  lane 0-15: M=lane, K=kb..kb+7 & kb+16..kb+23 (kb=kt*32);
// lane 16-31: same rows, kb += 8.
static __device__ inline v16h load_a(const _Float16* buf, int strideH, int kt, int lane) {
  int m  = lane & 15;
  int kb = kt * 32 + ((lane & 16) ? 8 : 0);
  const _Float16* p = buf + (size_t)m * strideH + kb;
  h8 lo = *(const h8*)p;
  h8 hi = *(const h8*)(p + 16);
  return make16(lo, hi);
}

// B fragment from pre-packed weights: frag = kt*ntiles+nt, 512 halves each.
static __device__ inline v16h load_b(const _Float16* pk, int frag, int lane) {
  const _Float16* p = pk + ((size_t)frag << 9) + lane * 16;
  h8 lo = *(const h8*)p;
  h8 hi = *(const h8*)(p + 8);
  return make16(lo, hi);
}

// single-tile matmul, B fragments fetched in chunks of CH so the scheduler can
// clause 2*CH b128 loads and issue CH wmmas per wait.
template<int KT, int NT, int CH>
static __device__ inline void mm1(const _Float16* __restrict__ aBuf, int strideH,
                                  const _Float16* __restrict__ wPack,
                                  v8f* c, int lane) {
  for (int kt = 0; kt < KT; ++kt) {
    v16h a = load_a(aBuf, strideH, kt, lane);
#pragma unroll
    for (int nc = 0; nc < NT; nc += CH) {
      v16h b[CH];
#pragma unroll
      for (int j = 0; j < CH; ++j)
        if (nc + j < NT) b[j] = load_b(wPack, kt * NT + nc + j, lane);
#pragma unroll
      for (int j = 0; j < CH; ++j)
        if (nc + j < NT)
          c[nc + j] = __builtin_amdgcn_wmma_f32_16x16x32_f16(
              false, a, false, b[j], (short)0, c[nc + j], false, false);
    }
  }
}

// dual-tile matmul: rows 0-15 (c0) and rows 16-31 (c1) share every B fragment.
template<int KT, int NT, int CH>
static __device__ inline void mm2(const _Float16* __restrict__ aBuf, int strideH,
                                  const _Float16* __restrict__ wPack,
                                  v8f* c0, v8f* c1, int lane) {
  for (int kt = 0; kt < KT; ++kt) {
    v16h a0 = load_a(aBuf, strideH, kt, lane);
    v16h a1 = load_a(aBuf + (size_t)16 * strideH, strideH, kt, lane);
#pragma unroll
    for (int nc = 0; nc < NT; nc += CH) {
      v16h b[CH];
#pragma unroll
      for (int j = 0; j < CH; ++j)
        if (nc + j < NT) b[j] = load_b(wPack, kt * NT + nc + j, lane);
#pragma unroll
      for (int j = 0; j < CH; ++j)
        if (nc + j < NT) {
          c0[nc + j] = __builtin_amdgcn_wmma_f32_16x16x32_f16(
              false, a0, false, b[j], (short)0, c0[nc + j], false, false);
          c1[nc + j] = __builtin_amdgcn_wmma_f32_16x16x32_f16(
              false, a1, false, b[j], (short)0, c1[nc + j], false, false);
        }
    }
  }
}

static __device__ inline float silu(float x) { return x / (1.f + __expf(-x)); }

// C/D layout: lane = N-col (0-15 twice), VGPR v = row (lanes>=16 -> rows 8-15).
static __device__ inline void store_act8(const v8f* c, const float* __restrict__ bias,
                                         _Float16* __restrict__ dst, int strideH,
                                         int lane, bool dosilu) {
  int n0 = lane & 15;
  int mh = (lane & 16) ? 8 : 0;
#pragma unroll
  for (int nt = 0; nt < 8; ++nt) {
    float bv = bias[nt * 16 + n0];
#pragma unroll
    for (int v = 0; v < 8; ++v) {
      float x = c[nt][v] + bv;
      if (dosilu) x = silu(x);
      dst[(size_t)(v + mh) * strideH + nt * 16 + n0] = (_Float16)x;
    }
  }
}

// ---------------- utility kernels ----------------
__global__ void zero_f32(float* p, long n) {
  long i = (long)blockIdx.x * blockDim.x + threadIdx.x;
  if (i < n) p[i] = 0.f;
}
__global__ void cvt_f32_f16(const float* __restrict__ s, _Float16* __restrict__ d, long n) {
  long i = (long)blockIdx.x * blockDim.x + threadIdx.x;
  if (i < n) d[i] = (_Float16)s[i];
}
// pack W[K x Nlog] (row-major, JAX x@W) into WMMA B fragments, padded to Npad cols
__global__ void pack_b(const float* __restrict__ W, _Float16* __restrict__ dst,
                       int K, int Nlog, int Npad) {
  int idx = blockIdx.x * blockDim.x + threadIdx.x;
  int total = K * Npad;
  if (idx >= total) return;
  int frag = idx >> 9, within = idx & 511;
  int lane = within >> 4, h = within & 15;
  int ntiles = Npad >> 4;
  int kt = frag / ntiles, nt = frag - kt * ntiles;
  int n  = (nt << 4) + (lane & 15);
  int kb = kt * 32 + ((lane & 16) ? 8 : 0);
  int k  = kb + ((h < 8) ? h : (8 + h));
  float v = (n < Nlog) ? W[(size_t)k * Nlog + n] : 0.f;
  dst[idx] = (_Float16)v;
}

// ---------------- edge pipeline: one wave handles 32 edges (2 row tiles) -----
struct EdgeArgs {
  const _Float16* nf16;
  const float* erad; const float* eang;
  const int* eidx;
  const float *b_eu0, *b_eu1, *b_eu2, *b_nu0, *b_nu1;
  const _Float16 *pk_eu0, *pk_eu1, *pk_eu2, *pk_nu0, *pk_nu1;
  float* agg; float* deg;
};

__global__ __launch_bounds__(32) void edge_kernel(EdgeArgs A) {
  __shared__ alignas(32) _Float16 sIn[32 * 384];  // [src | dst | ef] rows
  __shared__ alignas(32) _Float16 sH[32 * 256];   // hidden ping-pong
  const int lane = threadIdx.x;
  const int base = blockIdx.x * 32;

  // ---- stage: async global->LDS gathers for src/dst rows, VALU cvt for ef ---
  {
    const uint64_t nfbase  = (uint64_t)A.nf16;
    const uint32_t sInBase = (uint32_t)(uintptr_t)&sIn[0];
    int r = lane >> 1, p = lane & 1;
#pragma unroll
    for (int pass = 0; pass < 2; ++pass) {
      int rr = r + pass * 16;
      int e  = base + rr;
      int s  = A.eidx[e];
      int d  = A.eidx[E + e];
      uint32_t gS = (uint32_t)(s * 256 + p * 128);   // byte offsets in nf16
      uint32_t gD = (uint32_t)(d * 256 + p * 128);
      uint32_t lS = sInBase + (uint32_t)((rr * 384 + p * 64) * 2);
      uint32_t lD = sInBase + (uint32_t)((rr * 384 + 128 + p * 64) * 2);
#pragma unroll
      for (int j = 0; j < 8; ++j) {
        asm volatile("global_load_async_to_lds_b128 %0, %1, %2"
                     :: "v"(lS + j * 16), "v"(gS + j * 16), "s"(nfbase)
                     : "memory");
        asm volatile("global_load_async_to_lds_b128 %0, %1, %2"
                     :: "v"(lD + j * 16), "v"(gD + j * 16), "s"(nfbase)
                     : "memory");
      }
      const float* pf = (p ? A.eang : A.erad) + (size_t)e * 64;
      _Float16* qe = sIn + rr * 384 + 256 + p * 64;
#pragma unroll
      for (int j = 0; j < 64; ++j) qe[j] = (_Float16)pf[j];
    }
    asm volatile("s_wait_asynccnt 0x0" ::: "memory");
  }
  __syncthreads();

  v8f c0[8], c1[8];
#define ZERO_C                                                   \
  _Pragma("unroll") for (int nt = 0; nt < 8; ++nt)               \
    for (int i = 0; i < 8; ++i) { c0[nt][i] = 0.f; c1[nt][i] = 0.f; }

  // L1: 384 -> 128, SiLU
  ZERO_C;
  mm2<12, 8, 4>(sIn, 384, A.pk_eu0, c0, c1, lane);
  store_act8(c0, A.b_eu0, sH, 256, lane, true);
  store_act8(c1, A.b_eu0, sH + (size_t)16 * 256, 256, lane, true);
  __syncthreads();
  // L2: 128 -> 128, SiLU
  ZERO_C;
  mm2<4, 8, 4>(sH, 256, A.pk_eu1, c0, c1, lane);
  store_act8(c0, A.b_eu1, sH + 128, 256, lane, true);
  store_act8(c1, A.b_eu1, sH + (size_t)16 * 256 + 128, 256, lane, true);
  __syncthreads();
  // L3: 128 -> 128, + ef residual -> overwrite ef columns of sIn
  ZERO_C;
  mm2<4, 8, 4>(sH + 128, 256, A.pk_eu2, c0, c1, lane);
  {
    int n0 = lane & 15, mh = (lane & 16) ? 8 : 0;
#pragma unroll
    for (int nt = 0; nt < 8; ++nt) {
      float bv = A.b_eu2[nt * 16 + n0];
#pragma unroll
      for (int v = 0; v < 8; ++v) {
        _Float16* s0 = sIn + (size_t)(v + mh) * 384 + 256 + nt * 16 + n0;
        _Float16* s1 = sIn + (size_t)(16 + v + mh) * 384 + 256 + nt * 16 + n0;
        *s0 = (_Float16)(c0[nt][v] + bv + (float)(*s0));
        *s1 = (_Float16)(c1[nt][v] + bv + (float)(*s1));
      }
    }
  }
  __syncthreads();
  // L4: msg_in = sIn cols 128..383 (256 wide), -> 128, SiLU
  ZERO_C;
  mm2<8, 8, 4>(sIn + 128, 384, A.pk_nu0, c0, c1, lane);
  store_act8(c0, A.b_nu0, sH, 256, lane, true);
  store_act8(c1, A.b_nu0, sH + (size_t)16 * 256, 256, lane, true);
  __syncthreads();
  // L5: 128 -> 128, then scatter-add into agg[dst]
  ZERO_C;
  mm2<4, 8, 4>(sH, 256, A.pk_nu1, c0, c1, lane);
  {
    int n0 = lane & 15, mh = (lane & 16) ? 8 : 0;
    int dn0[8], dn1[8];
#pragma unroll
    for (int v = 0; v < 8; ++v) {
      dn0[v] = A.eidx[E + base + v + mh];
      dn1[v] = A.eidx[E + base + 16 + v + mh];
    }
#pragma unroll
    for (int nt = 0; nt < 8; ++nt) {
      float bv = A.b_nu1[nt * 16 + n0];
#pragma unroll
      for (int v = 0; v < 8; ++v) {
        atomicAdd(&A.agg[(size_t)dn0[v] * 128 + nt * 16 + n0], c0[nt][v] + bv);
        atomicAdd(&A.agg[(size_t)dn1[v] * 128 + nt * 16 + n0], c1[nt][v] + bv);
      }
    }
    atomicAdd(&A.deg[A.eidx[E + base + lane]], 1.0f);
  }
#undef ZERO_C
}

// ---------------- node finalize ----------------
__global__ void finalize_kernel(const float* __restrict__ agg, const float* __restrict__ deg,
                                const float* __restrict__ nf, _Float16* __restrict__ nfu16) {
  long i = (long)blockIdx.x * blockDim.x + threadIdx.x;
  if (i >= (long)N * 128) return;
  int n = (int)(i >> 7);
  float dg = fmaxf(deg[n], 1.0f);
  nfu16[i] = (_Float16)(agg[i] / dg + nf[i]);
}

// ---------------- routed extraction heads ----------------
struct HeadArgs {
  const _Float16* nfu16;
  const int* atom;
  const float* hb[6];        // [t*3 + layer]
  const _Float16* hpk[6];
  float* out;
};

__global__ __launch_bounds__(32) void heads_kernel(HeadArgs A) {
  __shared__ alignas(32) _Float16 sA[16 * 256];
  const int lane = threadIdx.x;
  const int base = blockIdx.x * 16;
  const int n0 = lane & 15, mh = (lane & 16) ? 8 : 0;
  const _Float16* aGlob = A.nfu16 + (size_t)base * 128;

  int at[8];
#pragma unroll
  for (int v = 0; v < 8; ++v) at[v] = A.atom[base + v + mh];

  for (int t = 0; t < 2; ++t) {
    v8f c[8];
#pragma unroll
    for (int nt = 0; nt < 8; ++nt) for (int i = 0; i < 8; ++i) c[nt][i] = 0.f;
    mm1<4, 8, 4>(aGlob, 128, A.hpk[t * 3 + 0], c, lane);
    store_act8(c, A.hb[t * 3 + 0], sA, 256, lane, true);
    __syncthreads();
#pragma unroll
    for (int nt = 0; nt < 8; ++nt) for (int i = 0; i < 8; ++i) c[nt][i] = 0.f;
    mm1<4, 8, 4>(sA, 256, A.hpk[t * 3 + 1], c, lane);
    store_act8(c, A.hb[t * 3 + 1], sA + 128, 256, lane, true);
    __syncthreads();
    v8f c2[11];
#pragma unroll
    for (int nt = 0; nt < 11; ++nt) for (int i = 0; i < 8; ++i) c2[nt][i] = 0.f;
    mm1<4, 11, 4>(sA + 128, 256, A.hpk[t * 3 + 2], c2, lane);
    const float* b2 = A.hb[t * 3 + 2];
#pragma unroll
    for (int nt = 0; nt < 11; ++nt) {
      int f = nt * 16 + n0;
      if (f < NO2) {
        float bv = b2[f];
#pragma unroll
        for (int v = 0; v < 8; ++v) {
          if (at[v] == t)
            A.out[(size_t)(base + v + mh) * NO2 + f] = c2[nt][v] + bv;
        }
      }
    }
    __syncthreads();
  }
}

// ---------------- host launch ----------------
extern "C" void kernel_launch(void* const* d_in, const int* in_sizes, int n_in,
                              void* d_out, int out_size, void* d_ws, size_t ws_size,
                              hipStream_t stream) {
  (void)in_sizes; (void)n_in; (void)out_size; (void)ws_size;
  // pytree order: nf, erad, eang, eidx, atom, then params sorted:
  // edge_update[0..2]{b,w}, heads[0][0..2]{b,w}, heads[1][0..2]{b,w}, node_update[0..1]{b,w}
  const float* nf    = (const float*)d_in[0];
  const float* erad  = (const float*)d_in[1];
  const float* eang  = (const float*)d_in[2];
  const int*   eidx  = (const int*)d_in[3];
  const int*   atom  = (const int*)d_in[4];
  const float* eu_b[3] = {(const float*)d_in[5],  (const float*)d_in[7],  (const float*)d_in[9]};
  const float* eu_w[3] = {(const float*)d_in[6],  (const float*)d_in[8],  (const float*)d_in[10]};
  const float* h_b[6]  = {(const float*)d_in[11], (const float*)d_in[13], (const float*)d_in[15],
                          (const float*)d_in[17], (const float*)d_in[19], (const float*)d_in[21]};
  const float* h_w[6]  = {(const float*)d_in[12], (const float*)d_in[14], (const float*)d_in[16],
                          (const float*)d_in[18], (const float*)d_in[20], (const float*)d_in[22]};
  const float* nu_b[2] = {(const float*)d_in[23], (const float*)d_in[25]};
  const float* nu_w[2] = {(const float*)d_in[24], (const float*)d_in[26]};

  char* ws = (char*)d_ws;
  size_t off = 0;
  auto alloc = [&](size_t bytes) { size_t r = off; off = (off + bytes + 255) & ~(size_t)255; return r; };
  float*     agg   = (float*)    (ws + alloc((size_t)N * 128 * 4));
  float*     deg   = (float*)    (ws + alloc((size_t)N * 4));
  _Float16*  nf16  = (_Float16*) (ws + alloc((size_t)N * 128 * 2));
  _Float16*  nfu16 = (_Float16*) (ws + alloc((size_t)N * 128 * 2));
  _Float16*  pk_eu0 = (_Float16*)(ws + alloc((size_t)384 * 128 * 2));
  _Float16*  pk_eu1 = (_Float16*)(ws + alloc((size_t)128 * 128 * 2));
  _Float16*  pk_eu2 = (_Float16*)(ws + alloc((size_t)128 * 128 * 2));
  _Float16*  pk_nu0 = (_Float16*)(ws + alloc((size_t)256 * 128 * 2));
  _Float16*  pk_nu1 = (_Float16*)(ws + alloc((size_t)128 * 128 * 2));
  _Float16*  pk_h[6];
  for (int i = 0; i < 6; ++i) {
    size_t cols = (i % 3 == 2) ? NO2P : 128;
    pk_h[i] = (_Float16*)(ws + alloc((size_t)128 * cols * 2));
  }

  // 1) zero accumulators
  {
    long n = (long)N * 128;
    zero_f32<<<(unsigned)((n + 255) / 256), 256, 0, stream>>>(agg, n);
    zero_f32<<<(unsigned)((N + 255) / 256), 256, 0, stream>>>(deg, N);
  }
  // 2) node features -> f16
  {
    long n = (long)N * 128;
    cvt_f32_f16<<<(unsigned)((n + 255) / 256), 256, 0, stream>>>(nf, nf16, n);
  }
  // 3) pack all weight matrices into WMMA B-fragment layout
  auto pack = [&](const float* W, _Float16* dst, int K, int Nlog, int Npad) {
    int total = K * Npad;
    pack_b<<<(total + 255) / 256, 256, 0, stream>>>(W, dst, K, Nlog, Npad);
  };
  pack(eu_w[0], pk_eu0, 384, 128, 128);
  pack(eu_w[1], pk_eu1, 128, 128, 128);
  pack(eu_w[2], pk_eu2, 128, 128, 128);
  pack(nu_w[0], pk_nu0, 256, 128, 128);
  pack(nu_w[1], pk_nu1, 128, 128, 128);
  for (int i = 0; i < 6; ++i) {
    int nl = (i % 3 == 2) ? NO2 : 128;
    int np = (i % 3 == 2) ? NO2P : 128;
    pack(h_w[i], pk_h[i], 128, nl, np);
  }
  // 4) edge pipeline (one wave per 32 edges)
  {
    EdgeArgs a;
    a.nf16 = nf16; a.erad = erad; a.eang = eang; a.eidx = eidx;
    a.b_eu0 = eu_b[0]; a.b_eu1 = eu_b[1]; a.b_eu2 = eu_b[2];
    a.b_nu0 = nu_b[0]; a.b_nu1 = nu_b[1];
    a.pk_eu0 = pk_eu0; a.pk_eu1 = pk_eu1; a.pk_eu2 = pk_eu2;
    a.pk_nu0 = pk_nu0; a.pk_nu1 = pk_nu1;
    a.agg = agg; a.deg = deg;
    edge_kernel<<<E / 32, 32, 0, stream>>>(a);
  }
  // 5) nf_upd = agg/max(deg,1) + nf  (f16)
  {
    long n = (long)N * 128;
    finalize_kernel<<<(unsigned)((n + 255) / 256), 256, 0, stream>>>(agg, deg, nf, nfu16);
  }
  // 6) routed heads (one wave per 16 nodes)
  {
    HeadArgs a;
    a.nfu16 = nfu16; a.atom = atom; a.out = (float*)d_out;
    for (int i = 0; i < 6; ++i) { a.hb[i] = h_b[i]; a.hpk[i] = pk_h[i]; }
    heads_kernel<<<N / 16, 32, 0, stream>>>(a);
  }
}
